// ForgetGates_14577119002753
// MI455X (gfx1250) — matrix operand order repaired and checked by reading, not compiled, not run on hardware
//
#include <hip/hip_runtime.h>
#include <hip/hip_bf16.h>

#define N_NODES 100000
#define N_EDGES 300000
#define HID     256
#define IN_DIM  1032
#define IN_PAD  1056   // 33*32, zero-padded K for W_if bf16 copy

typedef __attribute__((ext_vector_type(16))) __bf16 v16bf;
typedef __attribute__((ext_vector_type(8)))  float  v8f;

// Convert 4 float4s (16 fp32) into a v16bf A tile.
__device__ __forceinline__ v16bf cvt16_bf16(float4 a, float4 b, float4 c, float4 d) {
    v16bf r;
    r[0]  = (__bf16)a.x; r[1]  = (__bf16)a.y; r[2]  = (__bf16)a.z; r[3]  = (__bf16)a.w;
    r[4]  = (__bf16)b.x; r[5]  = (__bf16)b.y; r[6]  = (__bf16)b.z; r[7]  = (__bf16)b.w;
    r[8]  = (__bf16)c.x; r[9]  = (__bf16)c.y; r[10] = (__bf16)c.z; r[11] = (__bf16)c.w;
    r[12] = (__bf16)d.x; r[13] = (__bf16)d.y; r[14] = (__bf16)d.z; r[15] = (__bf16)d.w;
    return r;
}

__device__ __forceinline__ v16bf loadB(const __bf16* __restrict__ p) {
    return *reinterpret_cast<const v16bf*>(p);
}

// ---------------------------------------------------------------------------
// Kernel 0: zero the segment-sum output (harness poisons d_out with 0xAA).
// ---------------------------------------------------------------------------
__global__ void zero_out_kernel(float* __restrict__ out, int n) {
    int i = blockIdx.x * blockDim.x + threadIdx.x;
    if (i < n) out[i] = 0.0f;
}

// ---------------------------------------------------------------------------
// Kernel 1: one-shot weight conversion fp32 -> bf16 (weights are L2-resident).
// W_if is K-padded to 1056 with zeros so the GEMM tail needs no predicates.
// ---------------------------------------------------------------------------
__global__ void convert_weights_kernel(const float* __restrict__ Wif,
                                       const float* __restrict__ Whf,
                                       __bf16* __restrict__ wif_b,
                                       __bf16* __restrict__ whf_b) {
    int i = blockIdx.x * blockDim.x + threadIdx.x;
    if (i < HID * IN_PAD) {
        int r = i / IN_PAD, k = i - r * IN_PAD;
        wif_b[i] = (k < IN_DIM) ? (__bf16)Wif[(size_t)r * IN_DIM + k] : (__bf16)0.0f;
    }
    if (i < HID * HID) whf_b[i] = (__bf16)Whf[i];
}

// ---------------------------------------------------------------------------
// Kernel 2: Wx = x @ W_if^T  -> ws  [N_NODES, HID]
// One wave = 16 rows x all 256 cols (16 f32 accumulator tiles).
// 32 predicate-free K-steps + one zero-padded tail step.
// Software-pipelined: next-A raw floats prefetched over the 16 WMMAs,
// B tile double-buffered so each WMMA waits with >=1 load of slack.
// ---------------------------------------------------------------------------
__global__ void __launch_bounds__(128) wx_gemm_kernel(const float* __restrict__ x,
                                                      const __bf16* __restrict__ wif_b,
                                                      float* __restrict__ wx) {
    const int lane  = threadIdx.x & 31;
    const int wave  = threadIdx.x >> 5;
    const int mtile = blockIdx.x * 4 + wave;
    if (mtile >= N_NODES / 16) return;          // wave-uniform: EXEC stays full

    const int  nl = lane & 15;
    const bool hi = lane >= 16;
    const int  node_base = mtile * 16;
    const float* arow = x + (size_t)(node_base + nl) * IN_DIM;
    const int aoff = hi ? 16 : 0;               // A: lane-half K half
    const int boff = hi ? 16 : 0;               // B: lane-half K half
    const __bf16* brow = wif_b + (size_t)nl * IN_PAD + boff;

    v8f acc[16];
#pragma unroll
    for (int t = 0; t < 16; ++t)
#pragma unroll
        for (int g = 0; g < 8; ++g) acc[t][g] = 0.0f;

    // Preload A for kb = 0.
    const float4* qa0 = reinterpret_cast<const float4*>(arow + aoff);
    const float4* qa1 = reinterpret_cast<const float4*>(arow + 8 + aoff);
    v16bf A = cvt16_bf16(qa0[0], qa0[1], qa1[0], qa1[1]);

    // ---- 32 clean K-steps (K = 0..1023) ----
    for (int kb = 0; kb < 1024; kb += 32) {
        // Prefetch next-iteration A (raw fp32), converted after the WMMAs.
        float4 n0, n1, n2, n3;
        const bool more = (kb + 32) < 1024;
        if (more) {
            const float4* p0 = reinterpret_cast<const float4*>(arow + kb + 32 + aoff);
            const float4* p1 = reinterpret_cast<const float4*>(arow + kb + 40 + aoff);
            n0 = p0[0]; n1 = p0[1]; n2 = p1[0]; n3 = p1[1];
        }

        const __bf16* bk = brow + kb;
        v16bf Bc = loadB(bk);                   // B tile t = 0
#pragma unroll
        for (int t = 0; t < 16; ++t) {
            v16bf Bn;
            if (t < 15) Bn = loadB(bk + (size_t)(t + 1) * 16 * IN_PAD);
            acc[t] = __builtin_amdgcn_wmma_f32_16x16x32_bf16(
                false, A, false, Bc, (short)0, acc[t], false, false);
            if (t < 15) Bc = Bn;
        }
        if (more) A = cvt16_bf16(n0, n1, n2, n3);
    }

    // ---- tail K-step (K = 1024..1031 valid, rest zero) ----
    {
        const float4* q = reinterpret_cast<const float4*>(arow + 1024);
        float4 a = q[0];
        float4 b = q[1];
        const __bf16 z = (__bf16)0.0f;
        v16bf At;
        At[0] = hi ? z : (__bf16)a.x; At[1] = hi ? z : (__bf16)a.y;
        At[2] = hi ? z : (__bf16)a.z; At[3] = hi ? z : (__bf16)a.w;
        At[4] = hi ? z : (__bf16)b.x; At[5] = hi ? z : (__bf16)b.y;
        At[6] = hi ? z : (__bf16)b.z; At[7] = hi ? z : (__bf16)b.w;
#pragma unroll
        for (int i = 8; i < 16; ++i) At[i] = z;
        const __bf16* bk = brow + 1024;         // hi lanes read the zero pad
        v16bf Bc = loadB(bk);
#pragma unroll
        for (int t = 0; t < 16; ++t) {
            v16bf Bn;
            if (t < 15) Bn = loadB(bk + (size_t)(t + 1) * 16 * IN_PAD);
            acc[t] = __builtin_amdgcn_wmma_f32_16x16x32_bf16(
                false, At, false, Bc, (short)0, acc[t], false, false);
            if (t < 15) Bc = Bn;
        }
    }

    // C/D layout: lane = col N (per 16-lane half), VGPR g = row M (+8 for hi)
#pragma unroll
    for (int t = 0; t < 16; ++t)
#pragma unroll
        for (int g = 0; g < 8; ++g) {
            const int m = g + (hi ? 8 : 0);
            wx[(size_t)(node_base + m) * HID + t * 16 + nl] = acc[t][g];
        }
}

// ---------------------------------------------------------------------------
// Kernel 3: per 16-edge tile: Uh = h @ W_hf^T, then fused
//   f = sigmoid(Uh + b_f + Wx[parent]);  atomicAdd(out[parent], f * c)
// ---------------------------------------------------------------------------
__global__ void __launch_bounds__(128) edge_fused_kernel(const float* __restrict__ h,
                                                         const float* __restrict__ c,
                                                         const int*   __restrict__ parent,
                                                         const __bf16* __restrict__ whf_b,
                                                         const float* __restrict__ bfv,
                                                         const float* __restrict__ wx,
                                                         float* __restrict__ out) {
    const int lane  = threadIdx.x & 31;
    const int wave  = threadIdx.x >> 5;
    const int etile = blockIdx.x * 4 + wave;
    if (etile >= N_EDGES / 16) return;          // 18750 tiles exactly

    const int  nl = lane & 15;
    const bool hi = lane >= 16;
    const int  ebase = etile * 16;
    const float* arow = h + (size_t)(ebase + nl) * HID;
    const int aoff = hi ? 16 : 0;
    const int boff = hi ? 16 : 0;
    const __bf16* brow = whf_b + (size_t)nl * HID + boff;

    int par[8];
#pragma unroll
    for (int g = 0; g < 8; ++g) par[g] = parent[ebase + (hi ? 8 : 0) + g];

    v8f acc[16];
#pragma unroll
    for (int t = 0; t < 16; ++t)
#pragma unroll
        for (int g = 0; g < 8; ++g) acc[t][g] = 0.0f;

    // Preload A for kb = 0.
    const float4* qa0 = reinterpret_cast<const float4*>(arow + aoff);
    const float4* qa1 = reinterpret_cast<const float4*>(arow + 8 + aoff);
    v16bf A = cvt16_bf16(qa0[0], qa0[1], qa1[0], qa1[1]);

#pragma unroll
    for (int kb = 0; kb < HID; kb += 32) {      // 8 K-steps, no tail
        float4 n0, n1, n2, n3;
        const bool more = (kb + 32) < HID;
        if (more) {
            const float4* p0 = reinterpret_cast<const float4*>(arow + kb + 32 + aoff);
            const float4* p1 = reinterpret_cast<const float4*>(arow + kb + 40 + aoff);
            n0 = p0[0]; n1 = p0[1]; n2 = p1[0]; n3 = p1[1];
        }

        const __bf16* bk = brow + kb;
        v16bf Bc = loadB(bk);
#pragma unroll
        for (int t = 0; t < 16; ++t) {
            v16bf Bn;
            if (t < 15) Bn = loadB(bk + (size_t)(t + 1) * 16 * HID);
            acc[t] = __builtin_amdgcn_wmma_f32_16x16x32_bf16(
                false, A, false, Bc, (short)0, acc[t], false, false);
            if (t < 15) Bc = Bn;
        }
        if (more) A = cvt16_bf16(n0, n1, n2, n3);
    }

    // Fused epilogue: gate + gather + scaled atomic segment-sum
#pragma unroll
    for (int t = 0; t < 16; ++t) {
        const int   n  = t * 16 + nl;
        const float bv = bfv[n];
#pragma unroll
        for (int g = 0; g < 8; ++g) {
            const int   m  = (hi ? 8 : 0) + g;
            const float z  = acc[t][g] + bv + wx[(size_t)par[g] * HID + n];
            const float fg = 1.0f / (1.0f + __expf(-z));
            const float cv = c[(size_t)(ebase + m) * HID + n];
            __hip_atomic_fetch_add(&out[(size_t)par[g] * HID + n], fg * cv,
                                   __ATOMIC_RELAXED, __HIP_MEMORY_SCOPE_AGENT);
        }
    }
}

// ---------------------------------------------------------------------------
extern "C" void kernel_launch(void* const* d_in, const int* in_sizes, int n_in,
                              void* d_out, int out_size, void* d_ws, size_t ws_size,
                              hipStream_t stream) {
    const float* x      = (const float*)d_in[0];
    const float* h_flat = (const float*)d_in[1];
    const float* c_flat = (const float*)d_in[2];
    const int*   parent = (const int*)  d_in[3];
    const float* W_if   = (const float*)d_in[4];
    const float* W_hf   = (const float*)d_in[5];
    const float* b_f    = (const float*)d_in[6];
    float* out = (float*)d_out;

    // Workspace layout (all 32B aligned):
    //   [0)                  wx    : N_NODES*HID f32   = 102,400,000 B
    //   [102,400,000)        wif_b : HID*IN_PAD bf16   =     540,672 B
    //   [102,940,672)        whf_b : HID*HID bf16      =     131,072 B
    char* ws = (char*)d_ws;
    float*  wx    = (float*)ws;
    __bf16* wif_b = (__bf16*)(ws + (size_t)N_NODES * HID * 4);
    __bf16* whf_b = (__bf16*)(ws + (size_t)N_NODES * HID * 4 + (size_t)HID * IN_PAD * 2);

    // 1) zero segment-sum output
    zero_out_kernel<<<(out_size + 255) / 256, 256, 0, stream>>>(out, out_size);

    // 2) one-shot bf16 weight conversion
    convert_weights_kernel<<<(HID * IN_PAD + 255) / 256, 256, 0, stream>>>(
        W_if, W_hf, wif_b, whf_b);

    // 3) Wx = x @ W_if^T  (6250 m-tiles, 4 waves / 128-thread block)
    const int n_mtiles = N_NODES / 16;
    wx_gemm_kernel<<<(n_mtiles + 3) / 4, 128, 0, stream>>>(x, wif_b, wx);

    // 4) fused Uh GEMM + gate + gather + atomic segment-sum (18750 e-tiles)
    const int n_etiles = N_EDGES / 16;
    edge_fused_kernel<<<(n_etiles + 3) / 4, 128, 0, stream>>>(
        h_flat, c_flat, parent, whf_b, b_f, wx, out);
}
